// DualDeltaBlock_53369263620130
// MI455X (gfx1250) — matrix operand order repaired and checked by reading, not compiled, not run on hardware
//
#include <hip/hip_runtime.h>
#include <hip/hip_bf16.h>
#include <math.h>

// ---------------- problem constants ----------------
static constexpr int BB    = 2;
static constexpr int TT    = 2048;
static constexpr int DD    = 1024;
static constexpr int HH    = 8;
static constexpr int CCH   = 64;    // chunk length
static constexpr int HDIM  = 128;   // head dim
static constexpr int NCH   = 32;    // chunks per sequence
static constexpr int MROWS = BB * TT; // 4096 token rows

// ---------------- WMMA / TDM types ----------------
typedef __attribute__((ext_vector_type(16))) _Float16 v16h;
typedef __attribute__((ext_vector_type(8)))  _Float16 v8h;
typedef __attribute__((ext_vector_type(8)))  float    v8f;
typedef __attribute__((ext_vector_type(4)))  unsigned int u32x4;
typedef __attribute__((ext_vector_type(8)))  int      i32x8;
typedef __attribute__((ext_vector_type(4)))  int      i32x4;

#ifndef __has_builtin
#define __has_builtin(x) 0
#endif
#if __has_builtin(__builtin_amdgcn_tensor_load_to_lds) && \
    __has_builtin(__builtin_amdgcn_s_wait_tensorcnt)
#define USE_TDM 1
#else
#define USE_TDM 0
#endif

__device__ __forceinline__ v8f wmma_f16(v16h a, v16h b, v8f c) {
  // D = A(16x32 f16) * B(32x16 f16) + C(16x16 f32)
  return __builtin_amdgcn_wmma_f32_16x16x32_f16(
      false, a, false, b, (short)0, c, false, false);
}

// A fragment: logical A[m][k], row-major f16, ld elements per row.
__device__ __forceinline__ v16h frag_a(const _Float16* base, int ld, int lane,
                                       int m0, int k0) {
  int r  = m0 + (lane & 15);
  int hg = (lane >> 4) & 1;
  const _Float16* p = base + (size_t)r * ld + k0 + hg * 16;
  v8h lo = *(const v8h*)p;
  v8h hi = *(const v8h*)(p + 8);
  v16h a;
#pragma unroll
  for (int i = 0; i < 8; ++i) { a[i] = lo[i]; a[i + 8] = hi[i]; }
  return a;
}

// A fragment with per-row f32 scale (rk * de).
__device__ __forceinline__ v16h frag_a_scaled(const _Float16* base, int ld,
                                              int lane, int m0, int k0,
                                              const float* rowscale) {
  int r  = m0 + (lane & 15);
  int hg = (lane >> 4) & 1;
  float sc = rowscale[r];
  const _Float16* p = base + (size_t)r * ld + k0 + hg * 16;
  v8h lo = *(const v8h*)p;
  v8h hi = *(const v8h*)(p + 8);
  v16h a;
#pragma unroll
  for (int i = 0; i < 8; ++i) {
    a[i]     = (_Float16)((float)lo[i] * sc);
    a[i + 8] = (_Float16)((float)hi[i] * sc);
  }
  return a;
}

// Transposed+scaled A fragment: logical A[m][k] = src[k][m]*rowscale[k].
__device__ __forceinline__ v16h frag_a_tscaled(const _Float16* base, int ld,
                                               int lane, int m0, int k0,
                                               const float* rowscale) {
  int m  = m0 + (lane & 15);
  int hg = (lane >> 4) & 1;
  v16h a;
#pragma unroll
  for (int i = 0; i < 8; ++i) {
    int c0 = k0 + hg * 16 + i;
    int c1 = k0 + hg * 16 + 8 + i;
    a[i]     = (_Float16)((float)base[(size_t)c0 * ld + m] * rowscale[c0]);
    a[i + 8] = (_Float16)((float)base[(size_t)c1 * ld + m] * rowscale[c1]);
  }
  return a;
}

// B fragment from N-major storage: logical B[k][n] = W[n][k] (row-major NxK).
__device__ __forceinline__ v16h frag_b(const _Float16* base, int ld, int lane,
                                       int n0, int k0) {
  int nn = n0 + (lane & 15);
  int g  = (lane >> 4) & 1;
  const _Float16* p = base + (size_t)nn * ld + k0 + g * 16;
  v8h lo = *(const v8h*)p;
  v8h hi = *(const v8h*)(p + 8);
  v16h b;
#pragma unroll
  for (int i = 0; i < 8; ++i) { b[i] = lo[i]; b[i + 8] = hi[i]; }
  return b;
}

// B fragment from K-major (row-major KxN) f16 storage (strided gather).
__device__ __forceinline__ v16h frag_b_kn_h(const _Float16* base, int ld,
                                            int lane, int k0, int n0) {
  int nn = n0 + (lane & 15);
  int g  = (lane >> 4) & 1;
  v16h b;
#pragma unroll
  for (int i = 0; i < 16; ++i)
    b[i] = base[(size_t)(k0 + g * 16 + i) * ld + nn];
  return b;
}

// B fragment from K-major f32 storage with convert (LDS-resident state S).
__device__ __forceinline__ v16h frag_b_kn_f32(const float* base, int ld,
                                              int lane, int k0, int n0) {
  int nn = n0 + (lane & 15);
  int g  = (lane >> 4) & 1;
  v16h b;
#pragma unroll
  for (int i = 0; i < 16; ++i)
    b[i] = (_Float16)base[(size_t)(k0 + g * 16 + i) * ld + nn];
  return b;
}

__device__ __forceinline__ void store_c_f32(float* base, int ld, int lane,
                                            int m0, int n0, v8f c) {
  int col = n0 + (lane & 15);
  int rb  = m0 + ((lane >> 4) & 1) * 8;
#pragma unroll
  for (int r = 0; r < 8; ++r) base[(size_t)(rb + r) * ld + col] = c[r];
}
__device__ __forceinline__ void store_c_f16(_Float16* base, int ld, int lane,
                                            int m0, int n0, v8f c) {
  int col = n0 + (lane & 15);
  int rb  = m0 + ((lane >> 4) & 1) * 8;
#pragma unroll
  for (int r = 0; r < 8; ++r)
    base[(size_t)(rb + r) * ld + col] = (_Float16)c[r];
}

__device__ __forceinline__ float sigmoidf_(float z) {
  return 1.f / (1.f + __expf(-z));
}
__device__ __forceinline__ float softplusf_(float z) {
  return (z > 20.f) ? z : log1pf(__expf(z));
}

// ---------------- kernels ----------------

__global__ void cvt_f32_f16(const float* __restrict__ in,
                            _Float16* __restrict__ out, int n) {
  int i = blockIdx.x * 256 + threadIdx.x;
  if (i < n) out[i] = (_Float16)in[i];
}

// C[m][0..63+n0] = A @ W^T (+resid). Each wave: 16(M) x 64(N) with A-frag
// reused across 4 B tiles -> 4x the WMMA per A-load (HBM-bound GEMM wants reuse).
__global__ __launch_bounds__(256) void gemm_f16_wT(
    const _Float16* __restrict__ A, const _Float16* __restrict__ W,
    const float* __restrict__ resid, float* __restrict__ C,
    int M, int N, int K) {
  const int lane = threadIdx.x;
  const int m0 = (blockIdx.y * 8 + threadIdx.y) * 16;
  const int n0 = blockIdx.x * 64;
  v8f acc0 = {}, acc1 = {}, acc2 = {}, acc3 = {};
  for (int k0 = 0; k0 < K; k0 += 32) {
    if (k0 + 256 < K)  // speculative prefetch of the A stream
      __builtin_prefetch(A + (size_t)(m0 + (lane & 15)) * K + k0 + 256, 0, 1);
    v16h a  = frag_a(A, K, lane, m0, k0);
    v16h b0 = frag_b(W, K, lane, n0, k0);
    v16h b1 = frag_b(W, K, lane, n0 + 16, k0);
    v16h b2 = frag_b(W, K, lane, n0 + 32, k0);
    v16h b3 = frag_b(W, K, lane, n0 + 48, k0);
    acc0 = wmma_f16(a, b0, acc0);
    acc1 = wmma_f16(a, b1, acc1);
    acc2 = wmma_f16(a, b2, acc2);
    acc3 = wmma_f16(a, b3, acc3);
  }
  const int rb = m0 + ((lane >> 4) & 1) * 8;
  v8f accs[4] = { acc0, acc1, acc2, acc3 };
#pragma unroll
  for (int q = 0; q < 4; ++q) {
    int col = n0 + q * 16 + (lane & 15);
#pragma unroll
    for (int r = 0; r < 8; ++r) {
      size_t off = (size_t)(rb + r) * N + col;
      float v = accs[q][r];
      if (resid) v += resid[off];
      C[off] = v;
    }
  }
}

// 40 small per-token projections: gate/beta1/beta2 (sigmoid) and dec1/dec2.
__global__ void smallproj_kernel(
    const float* __restrict__ x,
    const float* __restrict__ Wg,  const float* __restrict__ Wb1,
    const float* __restrict__ Wb2, const float* __restrict__ Wa1,
    const float* __restrict__ Wa2, const float* __restrict__ dtb1,
    const float* __restrict__ dtb2, const float* __restrict__ Al1,
    const float* __restrict__ Al2,
    float* __restrict__ gate, float* __restrict__ b1, float* __restrict__ b2,
    float* __restrict__ d1, float* __restrict__ d2) {
  int m = blockIdx.x;
  int j = threadIdx.x;
  if (j >= 40) return;
  int grp = j >> 3, h = j & 7;
  const float* Wr =
      (grp == 0 ? Wg : grp == 1 ? Wb1 : grp == 2 ? Wb2 : grp == 3 ? Wa1 : Wa2)
      + (size_t)h * DD;
  const float* xr = x + (size_t)m * DD;
  float acc = 0.f;
  for (int k = 0; k < DD; ++k) acc += xr[k] * Wr[k];
  size_t o = (size_t)m * HH + h;
  if (grp == 0)      gate[o] = sigmoidf_(acc);
  else if (grp == 1) b1[o]   = sigmoidf_(acc);
  else if (grp == 2) b2[o]   = sigmoidf_(acc);
  else if (grp == 3) d1[o]   = -__expf(Al1[h]) * softplusf_(acc + dtb1[h]);
  else               d2[o]   = -__expf(Al2[h]) * softplusf_(acc + dtb2[h]);
}

// rk = xh / max(||xh||, 1e-12) per (token, head); store f16.
__global__ void rk_kernel(const float* __restrict__ x,
                          _Float16* __restrict__ rk16) {
  int m = blockIdx.x;
  int h = threadIdx.y;
  int lane = threadIdx.x;
  const float* xr = x + (size_t)m * DD + h * HDIM + lane * 4;
  float4 v = *(const float4*)xr;
  float ss = v.x * v.x + v.y * v.y + v.z * v.z + v.w * v.w;
#pragma unroll
  for (int msk = 16; msk; msk >>= 1) ss += __shfl_xor(ss, msk, 32);
  float sc = 1.f / fmaxf(sqrtf(ss), 1e-12f);
  _Float16* o = rk16 + (size_t)m * DD + h * HDIM + lane * 4;
  o[0] = (_Float16)(v.x * sc); o[1] = (_Float16)(v.y * sc);
  o[2] = (_Float16)(v.z * sc); o[3] = (_Float16)(v.w * sc);
}

// wk[t] = rk[t-1] (zero at t==0, per batch).
__global__ void wkshift_kernel(const _Float16* __restrict__ rk16,
                               _Float16* __restrict__ wk16) {
  int i = blockIdx.x * 256 + threadIdx.x;
  if (i >= MROWS * DD) return;
  int m = i >> 10;
  int t = m & (TT - 1);
  wk16[i] = (t == 0) ? (_Float16)0.f : rk16[i - DD];
}

// per-(b,h,n): cumsums c1/c2 and exp-derived arrays.
__global__ void decprep_kernel(
    const float* __restrict__ d1, const float* __restrict__ d2,
    float* __restrict__ c1g, float* __restrict__ c2g,
    float* __restrict__ de1, float* __restrict__ de2,
    float* __restrict__ dw1, float* __restrict__ dw2,
    float* __restrict__ lam1, float* __restrict__ lam2) {
  int tid = blockIdx.x * blockDim.x + threadIdx.x;
  if (tid >= BB * HH * NCH) return;
  int n = tid & (NCH - 1);
  int bh = tid >> 5;
  int h = bh % HH, b = bh / HH;
  int t0 = n * CCH;
  float s1 = 0.f, s2 = 0.f;
  for (int c = 0; c < CCH; ++c) {
    int t = t0 + c;
    size_t ii = (size_t)(b * TT + t) * HH + h;
    s1 += d1[ii]; s2 += d2[ii];
    c1g[(size_t)bh * TT + t] = s1;
    c2g[(size_t)bh * TT + t] = s2;
  }
  float l1 = s1, l2 = s2;
  for (int c = 0; c < CCH; ++c) {
    size_t o = (size_t)bh * TT + t0 + c;
    float a1 = c1g[o], a2 = c2g[o];
    de1[o] = __expf(a1); de2[o] = __expf(a2);
    dw1[o] = __expf(l1 - a1); dw2[o] = __expf(l2 - a2);
  }
  lam1[(size_t)bh * NCH + n] = __expf(l1);
  lam2[(size_t)bh * NCH + n] = __expf(l2);
}

// per-(b,h,n) chunk-local math.
__global__ __launch_bounds__(256) void chunk_kernel(
    const _Float16* __restrict__ rk16, const _Float16* __restrict__ wk16,
    const float* __restrict__ vals,
    const float* __restrict__ b1g, const float* __restrict__ b2g,
    const float* __restrict__ c1g, const float* __restrict__ c2g,
    const float* __restrict__ de1g, const float* __restrict__ de2g,
    _Float16* __restrict__ v1c16, _Float16* __restrict__ v2c16,
    _Float16* __restrict__ wc1c16, _Float16* __restrict__ wc2c16,
    _Float16* __restrict__ in1g, _Float16* __restrict__ in2g) {
  extern __shared__ __align__(32) char smem[];
  float*    rw  = (float*)smem;                   // 64x64 rk·wkᵀ
  float*    Mm  = rw + 64 * 64;                   // 64x64 M
  float*    Am  = Mm + 64 * 64;                   // 64x64 A (f32)
  _Float16* A16 = (_Float16*)(Am + 64 * 64);      // 64x64 A (f16)
  _Float16* stg = A16 + 64 * 64;                  // 64x128 staging (f16)
  float*    wwf = (float*)stg;                    // aliases stg: wk·wkᵀ
  float*    cav = (float*)(stg + 64 * HDIM);
  float* cc1 = cav + 64; float* cc2 = cc1 + 64;
  float* bb1 = cc2 + 64; float* bb2 = bb1 + 64;
  float* dl1 = bb2 + 64; float* dl2 = dl1 + 64;

  const int lane = threadIdx.x, w = threadIdx.y;
  const int tid = w * 32 + lane;
  const int n  = blockIdx.x % NCH;
  const int bh = blockIdx.x / NCH;
  const int h  = bh % HH, b = bh / HH;
  const int t0 = n * CCH;
  const size_t rowbase = (size_t)(b * TT + t0);
  const size_t rkoff   = rowbase * DD + (size_t)h * HDIM;
  const size_t bhT     = (size_t)bh * TT + t0;

  if (tid < 64) {
    float x1 = c1g[bhT + tid], x2 = c2g[bhT + tid];
    cc1[tid] = x1; cc2[tid] = x2; cav[tid] = 0.5f * (x1 + x2);
    size_t ib = (rowbase + tid) * HH + h;
    bb1[tid] = b1g[ib]; bb2[tid] = b2g[ib];
    dl1[tid] = de1g[bhT + tid]; dl2[tid] = de2g[bhT + tid];
  }
  __syncthreads();

  // Gram matrices: rw = rk·wkᵀ, ww = wk·wkᵀ (share B fragments).
  for (int tt = w; tt < 16; tt += 8) {
    int ti = (tt >> 2) * 16, tj = (tt & 3) * 16;
    v8f a1 = {}; v8f a2 = {};
#pragma unroll
    for (int k0 = 0; k0 < HDIM; k0 += 32) {
      v16h ar = frag_a(rk16 + rkoff, DD, lane, ti, k0);
      v16h aw = frag_a(wk16 + rkoff, DD, lane, ti, k0);
      v16h bw = frag_b(wk16 + rkoff, DD, lane, tj, k0);
      a1 = wmma_f16(ar, bw, a1);
      a2 = wmma_f16(aw, bw, a2);
    }
    store_c_f32(rw, 64, lane, ti, tj, a1);
    store_c_f32(wwf, 64, lane, ti, tj, a2);
  }
  __syncthreads();

  // M = tril(-(0.5*(b1+b2)[c] * ww[c][k] * exp(cavg[c]-cavg[k])), k=-1)
  for (int idx = tid; idx < 64 * 64; idx += 256) {
    int c = idx >> 6, k = idx & 63;
    float m = 0.f;
    if (c > k)
      m = -(0.5f * (bb1[c] + bb2[c]) * wwf[idx] * __expf(cav[c] - cav[k]));
    Mm[idx] = m;
  }
  __syncthreads();

  // Forward substitution (I - M) A = I; one column per thread, no cross-deps.
  if (tid < 64) {
    int j = tid;
    for (int i = 0; i < 64; ++i) {
      float s = 0.f;
      if (i == j) s = 1.f;
      else if (i > j) {
        for (int l = j; l < i; ++l) s += Mm[i * 64 + l] * Am[l * 64 + j];
      }
      Am[i * 64 + j] = s;
      A16[i * 64 + j] = (_Float16)s;
    }
  }
  // in1/in2 = tril(rkwk * L) — uses rw only, concurrent with the solve.
  for (int idx = tid; idx < 64 * 64; idx += 256) {
    int c = idx >> 6, k = idx & 63;
    float rv = rw[idx];
    _Float16 o1 = (_Float16)0.f, o2 = (_Float16)0.f;
    if (c >= k) {
      o1 = (_Float16)(rv * __expf(cc1[c] - cc1[k]));
      o2 = (_Float16)(rv * __expf(cc2[c] - cc2[k]));
    }
    in1g[(bhT + c) * CCH + k] = o1;
    in2g[(bhT + c) * CCH + k] = o2;
  }
  __syncthreads();

  // Four products: A(64x64) @ staged(64x128).
  _Float16* prodout[4] = { v1c16, v2c16, wc1c16, wc2c16 };
  for (int p = 0; p < 4; ++p) {
    for (int idx = tid; idx < 64 * HDIM; idx += 256) {
      int c = idx >> 7, d = idx & 127;
      float v;
      if (p == 0)
        v = vals[(rowbase + c) * (size_t)(2 * DD) + (size_t)h * HDIM + d] * bb1[c];
      else if (p == 1)
        v = vals[(rowbase + c) * (size_t)(2 * DD) + DD + (size_t)h * HDIM + d] * bb2[c];
      else if (p == 2)
        v = (float)wk16[(rowbase + c) * DD + (size_t)h * HDIM + d] * bb1[c] * dl1[c];
      else
        v = (float)wk16[(rowbase + c) * DD + (size_t)h * HDIM + d] * bb2[c] * dl2[c];
      stg[idx] = (_Float16)v;
    }
    __syncthreads();
    _Float16* op = prodout[p];
    for (int tt = w; tt < 32; tt += 8) {
      int ti = (tt >> 3) * 16, tj = (tt & 7) * 16;
      v8f acc = {};
#pragma unroll
      for (int k0 = 0; k0 < 64; k0 += 32) {
        v16h a = frag_a(A16, 64, lane, ti, k0);
        v16h bf = frag_b_kn_h(stg, HDIM, lane, k0, tj);
        acc = wmma_f16(a, bf, acc);
      }
      store_c_f16(op + bhT * HDIM, HDIM, lane, ti, tj, acc);
    }
    __syncthreads();
  }
}

// Sequential scan over chunks; S1/S2 (128x128 f32) in LDS; the per-chunk
// wc tile (16KB contiguous) is staged into LDS by the Tensor Data Mover.
__global__ __launch_bounds__(256) void scan_kernel(
    const _Float16* __restrict__ rk16, const _Float16* __restrict__ wk16,
    const _Float16* __restrict__ v1c16, const _Float16* __restrict__ v2c16,
    const _Float16* __restrict__ wc1c16, const _Float16* __restrict__ wc2c16,
    const _Float16* __restrict__ in1g, const _Float16* __restrict__ in2g,
    const float* __restrict__ de1g, const float* __restrict__ de2g,
    const float* __restrict__ dw1g, const float* __restrict__ dw2g,
    const float* __restrict__ lam1g, const float* __restrict__ lam2g,
    float* __restrict__ obuf) {
  extern __shared__ __align__(32) char smem[];
  float*    S1 = (float*)smem;
  float*    S2 = S1 + HDIM * HDIM;
  _Float16* vn  = (_Float16*)(S2 + HDIM * HDIM);  // 64x128 f16
  _Float16* wcl = vn + CCH * HDIM;                // 64x128 f16 (TDM dest)
  float* dloc  = (float*)(wcl + CCH * HDIM);
  float* dwloc = dloc + CCH;
  float* red   = dwloc + CCH;                     // 16 slots

  const int lane = threadIdx.x, w = threadIdx.y;
  const int tid = w * 32 + lane;
  const int bh = blockIdx.x;
  const int b = bh / HH, h = bh % HH;

  for (int idx = tid; idx < HDIM * HDIM; idx += 256) { S1[idx] = 0.f; S2[idx] = 0.f; }
  __syncthreads();

  for (int n = 0; n < NCH; ++n) {
    const int t0 = n * CCH;
    const size_t bhT = (size_t)bh * TT + t0;
    const size_t rkoff = ((size_t)(b * TT + t0)) * DD + (size_t)h * HDIM;

    for (int s = 0; s < 2; ++s) {
      float* S = s ? S2 : S1;
      const _Float16* vc = s ? v2c16 : v1c16;
      const _Float16* wc = s ? wc2c16 : wc1c16;
      const _Float16* inm = s ? in2g : in1g;
      const float* de = s ? de2g : de1g;
      const float* dw = s ? dw2g : dw1g;
      const float lam = (s ? lam2g : lam1g)[(size_t)bh * NCH + n];

      if (tid < CCH) {
        dloc[tid]  = de[bhT + tid];
        dwloc[tid] = dw[bhT + tid];
      }
#if USE_TDM
      if (w == 0) {
        // D# group0: count=1, lds_addr, global_addr, type=2 ("image")
        unsigned long long ga = (unsigned long long)(const void*)(wc + bhT * HDIM);
        unsigned lds_off = (unsigned)(unsigned long long)(void*)wcl;
        u32x4 g0;
        g0[0] = 1u;
        g0[1] = lds_off;
        g0[2] = (unsigned)ga;
        g0[3] = ((unsigned)(ga >> 32) & 0x01FFFFFFu) | (2u << 30);
        // D# group1: data_size=2B; 1-D tensor/tile of 8192 elements (16KB)
        i32x8 g1 = {};
        g1[0] = 0x00010000;  // data_size = 1 -> 2 bytes
        g1[1] = 0x20000000;  // tensor_dim0 = 8192 (low 16 bits << 16)
        g1[2] = 0x00010000;  // tensor_dim1 = 1
        g1[3] = 0x20000000;  // tile_dim0 = 8192
        g1[4] = 1;           // tile_dim1 = 1
        g1[5] = 8192;        // tensor_dim0_stride = 8192
        i32x4 z4 = {};
        i32x8 z8 = {};
        __builtin_amdgcn_tensor_load_to_lds(g0, g1, z4, z4, z8, 0);
        __builtin_amdgcn_s_wait_tensorcnt(0);
      }
#else
      for (int idx = tid; idx < CCH * HDIM; idx += 256)
        wcl[idx] = wc[bhT * HDIM + idx];
#endif
      __syncthreads();
      // (a) vn = vc - wc @ S   (64x128, K=128); wc read from LDS copy
      for (int tt = w; tt < 32; tt += 8) {
        int ti = (tt >> 3) * 16, tj = (tt & 7) * 16;
        v8f acc = {};
#pragma unroll
        for (int k0 = 0; k0 < HDIM; k0 += 32) {
          v16h a = frag_a(wcl, HDIM, lane, ti, k0);
          v16h bf = frag_b_kn_f32(S, HDIM, lane, k0, tj);
          acc = wmma_f16(a, bf, acc);
        }
        int col = tj + (lane & 15);
        int rb  = ti + ((lane >> 4) & 1) * 8;
#pragma unroll
        for (int r = 0; r < 8; ++r) {
          int c = rb + r;
          float v = (float)vc[(bhT + c) * HDIM + col] - acc[r];
          vn[c * HDIM + col] = (_Float16)v;
        }
      }
      __syncthreads();
      // (b) o += (rk*de)@S + in@vn   (64x128)
      for (int tt = w; tt < 32; tt += 8) {
        int ti = (tt >> 3) * 16, tj = (tt & 7) * 16;
        v8f acc = {};
#pragma unroll
        for (int k0 = 0; k0 < HDIM; k0 += 32) {
          v16h a = frag_a_scaled(rk16 + rkoff, DD, lane, ti, k0, dloc);
          v16h bf = frag_b_kn_f32(S, HDIM, lane, k0, tj);
          acc = wmma_f16(a, bf, acc);
        }
#pragma unroll
        for (int k0 = 0; k0 < CCH; k0 += 32) {
          v16h a = frag_a(inm + bhT * CCH, CCH, lane, ti, k0);
          v16h bf = frag_b_kn_h(vn, HDIM, lane, k0, tj);
          acc = wmma_f16(a, bf, acc);
        }
        int col = tj + (lane & 15);
        int rb  = ti + ((lane >> 4) & 1) * 8;
#pragma unroll
        for (int r = 0; r < 8; ++r) {
          size_t off = (bhT + rb + r) * HDIM + col;
          obuf[off] = (s == 0) ? acc[r] : (obuf[off] + acc[r]);
        }
      }
      __syncthreads();
      // (c) S = S*lam + (wk*dw)ᵀ @ vn   (128x128, K=64); accumulate ||S||²
      float ss = 0.f;
      for (int tt = w; tt < 64; tt += 8) {
        int ti = (tt >> 3) * 16, tj = (tt & 7) * 16;
        v8f acc = {};
#pragma unroll
        for (int k0 = 0; k0 < CCH; k0 += 32) {
          v16h a = frag_a_tscaled(wk16 + rkoff, DD, lane, ti, k0, dwloc);
          v16h bf = frag_b_kn_h(vn, HDIM, lane, k0, tj);
          acc = wmma_f16(a, bf, acc);
        }
        int col = tj + (lane & 15);
        int rb  = ti + ((lane >> 4) & 1) * 8;
#pragma unroll
        for (int r = 0; r < 8; ++r) {
          int d = rb + r;
          float v = S[d * HDIM + col] * lam + acc[r];
          S[d * HDIM + col] = v;
          ss += v * v;
        }
      }
#pragma unroll
      for (int msk = 16; msk; msk >>= 1) ss += __shfl_xor(ss, msk, 32);
      if (lane == 0) red[w] = ss;
      __syncthreads();
      if (tid == 0) {
        float tot = 0.f;
        for (int i = 0; i < 8; ++i) tot += red[i];
        float nrm = sqrtf(tot);
        red[8] = fminf(nrm, 100.f) / fmaxf(nrm, 1e-6f);
      }
      __syncthreads();
      float scl = red[8];
      for (int idx = tid; idx < HDIM * HDIM; idx += 256) S[idx] *= scl;
      __syncthreads();
    }
  }
}

// o (B,H,T,HD) * gate -> (B,T,D) f16 for the final projection.
__global__ void gate_kernel(const float* __restrict__ obuf,
                            const float* __restrict__ gate,
                            _Float16* __restrict__ og16) {
  int i = blockIdx.x * 256 + threadIdx.x;
  if (i >= MROWS * DD) return;
  int m = i >> 10, d = i & 1023;
  int h = d >> 7, dd = d & 127;
  int b = m >> 11, t = m & (TT - 1);
  float v = obuf[((size_t)(b * HH + h) * TT + t) * HDIM + dd] *
            gate[(size_t)m * HH + h];
  og16[i] = (_Float16)v;
}

// ---------------- host launcher ----------------
extern "C" void kernel_launch(void* const* d_in, const int* in_sizes, int n_in,
                              void* d_out, int out_size, void* d_ws,
                              size_t ws_size, hipStream_t stream) {
  (void)in_sizes; (void)n_in; (void)out_size; (void)ws_size;
  const float* x    = (const float*)d_in[0];
  const float* Wwr  = (const float*)d_in[1];
  const float* Wg   = (const float*)d_in[2];
  const float* Wo   = (const float*)d_in[3];
  const float* Wb1  = (const float*)d_in[4];
  const float* Wb2  = (const float*)d_in[5];
  const float* Wa1  = (const float*)d_in[6];
  const float* Wa2  = (const float*)d_in[7];
  const float* dtb1 = (const float*)d_in[8];
  const float* dtb2 = (const float*)d_in[9];
  const float* Al1  = (const float*)d_in[10];
  const float* Al2  = (const float*)d_in[11];
  float* outp = (float*)d_out;

  char* ws = (char*)d_ws;
  size_t off = 0;
  auto carve = [&](size_t bytes) -> char* {
    char* p = ws + off;
    off += (bytes + 255) & ~(size_t)255;
    return p;
  };

  _Float16* x16  = (_Float16*)carve((size_t)MROWS * DD * 2);
  _Float16* ww16 = (_Float16*)carve((size_t)2 * DD * DD * 2);
  _Float16* wo16 = (_Float16*)carve((size_t)DD * DD * 2);
  float* vals = (float*)carve((size_t)MROWS * 2 * DD * 4);
  float* gate = (float*)carve((size_t)MROWS * HH * 4);
  float* b1s  = (float*)carve((size_t)MROWS * HH * 4);
  float* b2s  = (float*)carve((size_t)MROWS * HH * 4);
  float* d1s  = (float*)carve((size_t)MROWS * HH * 4);
  float* d2s  = (float*)carve((size_t)MROWS * HH * 4);
  _Float16* rk16 = (_Float16*)carve((size_t)MROWS * DD * 2);
  _Float16* wk16 = (_Float16*)carve((size_t)MROWS * DD * 2);
  float* c1g = (float*)carve((size_t)BB * HH * TT * 4);
  float* c2g = (float*)carve((size_t)BB * HH * TT * 4);
  float* de1 = (float*)carve((size_t)BB * HH * TT * 4);
  float* de2 = (float*)carve((size_t)BB * HH * TT * 4);
  float* dw1 = (float*)carve((size_t)BB * HH * TT * 4);
  float* dw2 = (float*)carve((size_t)BB * HH * TT * 4);
  float* lam1 = (float*)carve((size_t)BB * HH * NCH * 4);
  float* lam2 = (float*)carve((size_t)BB * HH * NCH * 4);
  _Float16* v1c = (_Float16*)carve((size_t)BB * HH * TT * HDIM * 2);
  _Float16* v2c = (_Float16*)carve((size_t)BB * HH * TT * HDIM * 2);
  _Float16* wc1 = (_Float16*)carve((size_t)BB * HH * TT * HDIM * 2);
  _Float16* wc2 = (_Float16*)carve((size_t)BB * HH * TT * HDIM * 2);
  _Float16* in1 = (_Float16*)carve((size_t)BB * HH * TT * CCH * 2);
  _Float16* in2 = (_Float16*)carve((size_t)BB * HH * TT * CCH * 2);
  float* obuf = (float*)carve((size_t)BB * HH * TT * HDIM * 4);
  _Float16* og16 = (_Float16*)carve((size_t)MROWS * DD * 2);

  const int nx = MROWS * DD;
  cvt_f32_f16<<<(nx + 255) / 256, 256, 0, stream>>>(x, x16, nx);
  const int nw = 2 * DD * DD;
  cvt_f32_f16<<<(nw + 255) / 256, 256, 0, stream>>>(Wwr, ww16, nw);
  const int no = DD * DD;
  cvt_f32_f16<<<(no + 255) / 256, 256, 0, stream>>>(Wo, wo16, no);

  // vals = x @ W_write.T   (4096 x 2048 x 1024), 16x64 per wave
  gemm_f16_wT<<<dim3(2 * DD / 64, MROWS / 128), dim3(32, 8), 0, stream>>>(
      x16, ww16, nullptr, vals, MROWS, 2 * DD, DD);

  smallproj_kernel<<<MROWS, 64, 0, stream>>>(x, Wg, Wb1, Wb2, Wa1, Wa2, dtb1,
                                             dtb2, Al1, Al2, gate, b1s, b2s,
                                             d1s, d2s);

  rk_kernel<<<MROWS, dim3(32, 8), 0, stream>>>(x, rk16);
  wkshift_kernel<<<(nx + 255) / 256, 256, 0, stream>>>(rk16, wk16);
  decprep_kernel<<<2, 256, 0, stream>>>(d1s, d2s, c1g, c2g, de1, de2, dw1, dw2,
                                        lam1, lam2);

  const size_t chunkSmem = 3 * 64 * 64 * sizeof(float)
                         + 64 * 64 * sizeof(_Float16)
                         + 64 * HDIM * sizeof(_Float16)
                         + 7 * 64 * sizeof(float);
  (void)hipFuncSetAttribute(reinterpret_cast<const void*>(chunk_kernel),
                            hipFuncAttributeMaxDynamicSharedMemorySize,
                            (int)chunkSmem);
  chunk_kernel<<<BB * HH * NCH, dim3(32, 8), chunkSmem, stream>>>(
      rk16, wk16, vals, b1s, b2s, c1g, c2g, de1, de2, v1c, v2c, wc1, wc2, in1,
      in2);

  const size_t scanSmem = 2 * HDIM * HDIM * sizeof(float)      // S1, S2
                        + 2 * CCH * HDIM * sizeof(_Float16)    // vn + wcl
                        + (CCH + CCH + 16) * sizeof(float);    // de, dw, red
  (void)hipFuncSetAttribute(reinterpret_cast<const void*>(scan_kernel),
                            hipFuncAttributeMaxDynamicSharedMemorySize,
                            (int)scanSmem);
  scan_kernel<<<BB * HH, dim3(32, 8), scanSmem, stream>>>(
      rk16, wk16, v1c, v2c, wc1, wc2, in1, in2, de1, de2, dw1, dw2, lam1, lam2,
      obuf);

  gate_kernel<<<(nx + 255) / 256, 256, 0, stream>>>(obuf, gate, og16);

  // out = residual + og @ W_out.T   (4096 x 1024 x 1024)
  gemm_f16_wT<<<dim3(DD / 64, MROWS / 128), dim3(32, 8), 0, stream>>>(
      og16, wo16, x, outp, MROWS, DD, DD);
}
